// DeliveryEventEncoder_34050500722767
// MI455X (gfx1250) — compile-verified, matrix operand order checked by reading, not run on hardware
//
#include <hip/hip_runtime.h>
#include <hip/hip_bf16.h>

#define B_    32
#define U_    32
#define L_    256
#define DSEQ_ 5
#define H_    128
#define DOUT_ 128
#define AGGD_ 7
#define TODD_ 3
#define UNITD_ 16
#define FUSED_ 26

typedef __attribute__((ext_vector_type(16))) _Float16 v16h;
typedef __attribute__((ext_vector_type(8)))  _Float16 v8h;
typedef __attribute__((ext_vector_type(8)))  float    v8f;

#define V8F0 {0.f,0.f,0.f,0.f,0.f,0.f,0.f,0.f}

__device__ __forceinline__ v8f wmma16(v16h a, v16h b, v8f c) {
  // D = A(16x32 f16) * B(32x16 f16) + C(16x16 f32)
  return __builtin_amdgcn_wmma_f32_16x16x32_f16(false, a, false, b, (short)0, c, false, false);
}

// Load a v16h A/B fragment from a row-major f16 tile (LDS or global).
// Caller passes p0 = base + row*stride + kchunk + ((lane<16)?0:8).
// halves 0..7 <- p0[0..7], halves 8..15 <- p0[16..23]  (16-bit WMMA K striping)
__device__ __forceinline__ v16h frag_f16(const _Float16* p0) {
  v16h r;
  ((v8h*)&r)[0] = *(const v8h*)(p0);
  ((v8h*)&r)[1] = *(const v8h*)(p0 + 16);
  return r;
}

// wave32: reduce within each 16-lane half (rows of a C-tile live in lane halves)
__device__ __forceinline__ float rsum16(float v) {
#pragma unroll
  for (int m = 1; m < 16; m <<= 1) v += __shfl_xor(v, m, 32);
  return v;
}
__device__ __forceinline__ float rmax16(float v) {
#pragma unroll
  for (int m = 1; m < 16; m <<= 1) v = fmaxf(v, __shfl_xor(v, m, 32));
  return v;
}

// LayerNorm over H=128 for a row-tile held as 8 C-layout accumulators.
__device__ __forceinline__ void layernorm_c(v8f xc[8], const float* __restrict__ w,
                                            const float* __restrict__ b, int c16) {
  float mu[8], iv[8];
#pragma unroll
  for (int i = 0; i < 8; ++i) {
    float s = 0.f;
#pragma unroll
    for (int t = 0; t < 8; ++t) s += xc[t][i];
    mu[i] = rsum16(s) * (1.0f / 128.0f);
  }
#pragma unroll
  for (int i = 0; i < 8; ++i) {
    float s = 0.f;
#pragma unroll
    for (int t = 0; t < 8; ++t) { float d = xc[t][i] - mu[i]; s += d * d; }
    iv[i] = rsqrtf(rsum16(s) * (1.0f / 128.0f) + 1e-5f);
  }
#pragma unroll
  for (int t = 0; t < 8; ++t) {
    float ww = w[t * 16 + c16], bb = b[t * 16 + c16];
#pragma unroll
    for (int i = 0; i < 8; ++i) xc[t][i] = (xc[t][i] - mu[i]) * iv[i] * ww + bb;
  }
}

// One-shot weight conversion f32 -> f16 (weights shared by all 1024 WGs).
__global__ __launch_bounds__(256)
void cvt_f16_kernel(const float* __restrict__ src, _Float16* __restrict__ dst, int n) {
  int i = blockIdx.x * blockDim.x + threadIdx.x;
  if (i < n) dst[i] = (_Float16)src[i];
}

// W_in [128][5] f32 -> f16 [128][32] zero-padded K panel (so emb runs on WMMA).
__global__ __launch_bounds__(256)
void pad_win_kernel(const float* __restrict__ W_in, _Float16* __restrict__ dst) {
  int i = blockIdx.x * blockDim.x + threadIdx.x;  // 128*32 = 4096
  if (i < H_ * 32) {
    int col = i >> 5, d = i & 31;
    dst[i] = (d < DSEQ_) ? (_Float16)W_in[col * DSEQ_ + d] : (_Float16)0.f;
  }
}

// Build the zero-padded x_seq A-fragment for one 16-row tile (K=5 in halves 0..4
// of lanes 0..15; everything else zero per the 16-bit A layout).
__device__ __forceinline__ v16h xseq_frag(const float* __restrict__ x_seq, int bu, int rt,
                                          int lane, int c16) {
  v16h a;
#pragma unroll
  for (int e = 0; e < 16; ++e) a[e] = (_Float16)0.f;
  if (lane < 16) {
    const float* xp = x_seq + ((size_t)bu * L_ + rt * 16 + c16) * DSEQ_;
#pragma unroll
    for (int d = 0; d < DSEQ_; ++d) a[d] = (_Float16)xp[d];
  }
  return a;
}

__global__ __launch_bounds__(256)
void encoder_kernel(const float* __restrict__ x_seq,
                    const _Float16* __restrict__ win_h,       // [128][32] f16, K zero-padded
                    const float* __restrict__ b_in,
                    const _Float16* __restrict__ in_proj_h,   // [384][128] f16
                    const float* __restrict__ in_proj_b,
                    const _Float16* __restrict__ out_proj_h,  // [128][128] f16
                    const float* __restrict__ out_proj_b, const float* __restrict__ ln1_w,
                    const float* __restrict__ ln1_b,
                    const _Float16* __restrict__ ff1_h,       // [128][128] f16
                    const float* __restrict__ b_ff1,
                    const _Float16* __restrict__ ff2_h,       // [128][128] f16
                    const float* __restrict__ b_ff2, const float* __restrict__ ln2_w,
                    const float* __restrict__ ln2_b, const float* __restrict__ W_unit,
                    const float* __restrict__ b_unit, const int* __restrict__ lengths,
                    float* __restrict__ unit_out) {
  // CDNA5: 320KB LDS per WGP -> Q, K and V^T all fully resident per unit.
  __shared__ _Float16 sQ[L_ * H_];          // 64 KB  Q,  row-major [l][h]
  __shared__ _Float16 sK[L_ * H_];          // 64 KB  K,  row-major [l][h]
  __shared__ _Float16 sVT[H_ * L_];         // 64 KB  V^T row-major [h][l]
  __shared__ _Float16 sStage[8 * 16 * H_];  // 32 KB  per-wave [16][128] staging
  __shared__ _Float16 sP[8 * 16 * 32];      //  8 KB  per-wave softmax P tiles
  __shared__ float    sPool[H_];

  const int bu    = blockIdx.x;           // (b,u) unit
  const int lane  = threadIdx.x & 31;
  const int wave  = threadIdx.x >> 5;     // 8 waves, 2 row-tiles each
  const int c16   = lane & 15;
  const int half8 = (lane < 16) ? 0 : 8;  // C-layout row offset for upper lanes
  const int boff  = (lane < 16) ? 0 : 8;  // fragment K striping offset
  const int len   = lengths[bu];

  _Float16* stg = sStage + wave * (16 * H_);
  _Float16* sp  = sP + wave * (16 * 32);

  if (threadIdx.x < H_) sPool[threadIdx.x] = 0.f;

  __builtin_prefetch(in_proj_h, 0, 1);
  __builtin_prefetch(out_proj_h, 0, 1);
  __builtin_prefetch(ff1_h, 0, 1);
  __builtin_prefetch(ff2_h, 0, 1);

  // ---------------- Phase A: emb -> Q, K, V^T (all LDS) ----------------
#pragma unroll 1
  for (int rti = 0; rti < 2; ++rti) {
    const int rt = wave * 2 + rti;
    const v16h axs = xseq_frag(x_seq, bu, rt, lane, c16);
    // emb tile via WMMA (one K=32 step per 16-col tile) -> staging (f16)
#pragma unroll 1
    for (int ot = 0; ot < 8; ++ot) {
      v8f ce = V8F0;
      ce = wmma16(axs, frag_f16(win_h + (ot * 16 + c16) * 32 + boff), ce);
      const int col = ot * 16 + c16;
      const float bi = b_in[col];
#pragma unroll
      for (int i = 0; i < 8; ++i)
        stg[(half8 + i) * H_ + col] = (_Float16)(ce[i] + bi);
    }
    v16h ea[4];
#pragma unroll
    for (int kk = 0; kk < 4; ++kk)
      ea[kk] = frag_f16(stg + c16 * H_ + kk * 32 + boff);

#pragma unroll 1
    for (int ot = 0; ot < 8; ++ot) {
      v8f cq = V8F0, ck = V8F0, cv = V8F0;
#pragma unroll
      for (int kk = 0; kk < 4; ++kk) {
        const int kb = kk * 32 + boff;
        cq = wmma16(ea[kk], frag_f16(in_proj_h + (0 * H_ + ot * 16 + c16) * H_ + kb), cq);
        ck = wmma16(ea[kk], frag_f16(in_proj_h + (1 * H_ + ot * 16 + c16) * H_ + kb), ck);
        cv = wmma16(ea[kk], frag_f16(in_proj_h + (2 * H_ + ot * 16 + c16) * H_ + kb), cv);
      }
      const int col = ot * 16 + c16;
      const float qb  = in_proj_b[col];
      const float kb2 = in_proj_b[H_ + col];
      const float vb  = in_proj_b[2 * H_ + col];
#pragma unroll
      for (int i = 0; i < 8; ++i) {
        const int r = rt * 16 + half8 + i;
        sQ[r * H_ + col]  = (_Float16)(cq[i] + qb);
        sK[r * H_ + col]  = (_Float16)(ck[i] + kb2);
        sVT[col * L_ + r] = (_Float16)(cv[i] + vb);  // transposed store
      }
    }
  }
  __syncthreads();  // all Q/K/V ready

  // ---------------- Phase B: attention + out_proj + LN + FFN + LN + pool ----------------
  const float scale = 0.08838834764831845f;  // 1/sqrt(128)
#pragma unroll 1
  for (int rti = 0; rti < 2; ++rti) {
    const int rt = wave * 2 + rti;
    // Q A-fragments: loop-local, constant indices
    v16h qa[4];
#pragma unroll
    for (int kk = 0; kk < 4; ++kk)
      qa[kk] = frag_f16(sQ + (rt * 16 + c16) * H_ + kk * 32 + boff);

    v8f Oc[8];
    float m[8], lsum[8];
#pragma unroll
    for (int t = 0; t < 8; ++t) {
#pragma unroll
      for (int i = 0; i < 8; ++i) Oc[t][i] = 0.f;
    }
#pragma unroll
    for (int i = 0; i < 8; ++i) { m[i] = -1e30f; lsum[i] = 0.f; }

#pragma unroll 1
    for (int kp = 0; kp < 8; ++kp) {  // key pairs: keys [kp*32, kp*32+32)
      v8f c0 = V8F0, c1 = V8F0;
#pragma unroll
      for (int kk = 0; kk < 4; ++kk) {
        const int kb = kk * 32 + boff;
        c0 = wmma16(qa[kk], frag_f16(sK + (kp * 32 + c16) * H_ + kb), c0);
        c1 = wmma16(qa[kk], frag_f16(sK + (kp * 32 + 16 + c16) * H_ + kb), c1);
      }
      const int k0 = kp * 32 + c16;
      const int k1 = k0 + 16;
#pragma unroll
      for (int i = 0; i < 8; ++i) {
        float s0 = (k0 < len) ? c0[i] * scale : -1e9f;
        float s1 = (k1 < len) ? c1[i] * scale : -1e9f;
        float nm = fmaxf(rmax16(fmaxf(s0, s1)), m[i]);
        float p0 = __expf(s0 - nm);
        float p1 = __expf(s1 - nm);
        float f  = __expf(m[i] - nm);
        lsum[i] = lsum[i] * f + rsum16(p0 + p1);
        m[i] = nm;
#pragma unroll
        for (int t = 0; t < 8; ++t) Oc[t][i] *= f;
        sp[(half8 + i) * 32 + c16]      = (_Float16)p0;
        sp[(half8 + i) * 32 + 16 + c16] = (_Float16)p1;
      }
      const v16h pa = frag_f16(sp + c16 * 32 + boff);
#pragma unroll
      for (int t = 0; t < 8; ++t)
        Oc[t] = wmma16(pa, frag_f16(sVT + (t * 16 + c16) * L_ + kp * 32 + boff), Oc[t]);
    }
#pragma unroll
    for (int i = 0; i < 8; ++i) {
      const float r = 1.f / lsum[i];
#pragma unroll
      for (int t = 0; t < 8; ++t) Oc[t][i] *= r;
    }

    // out_proj: stage O as f16, reload A-fragments
#pragma unroll
    for (int t = 0; t < 8; ++t)
#pragma unroll
      for (int i = 0; i < 8; ++i)
        stg[(half8 + i) * H_ + t * 16 + c16] = (_Float16)Oc[t][i];
    v16h oa[4];
#pragma unroll
    for (int kk = 0; kk < 4; ++kk)
      oa[kk] = frag_f16(stg + c16 * H_ + kk * 32 + boff);

    v8f xc[8];
#pragma unroll 1
    for (int t = 0; t < 8; ++t) {
      v8f acc = V8F0;
#pragma unroll
      for (int kk = 0; kk < 4; ++kk)
        acc = wmma16(oa[kk], frag_f16(out_proj_h + (t * 16 + c16) * H_ + kk * 32 + boff), acc);
      const float ob = out_proj_b[t * 16 + c16];
#pragma unroll
      for (int i = 0; i < 8; ++i) xc[t][i] = acc[i] + ob;
    }

    // residual: emb recomputed on WMMA from a freshly built x_seq fragment
    {
      const v16h axs = xseq_frag(x_seq, bu, rt, lane, c16);
#pragma unroll 1
      for (int t = 0; t < 8; ++t) {
        v8f ce = V8F0;
        ce = wmma16(axs, frag_f16(win_h + (t * 16 + c16) * 32 + boff), ce);
        const float bi = b_in[t * 16 + c16];
#pragma unroll
        for (int i = 0; i < 8; ++i) xc[t][i] += ce[i] + bi;
      }
    }
    layernorm_c(xc, ln1_w, ln1_b, c16);

    // FFN: stage x, FF1 (relu), stage ff, FF2 + residual, LN2
#pragma unroll
    for (int t = 0; t < 8; ++t)
#pragma unroll
      for (int i = 0; i < 8; ++i)
        stg[(half8 + i) * H_ + t * 16 + c16] = (_Float16)xc[t][i];
    v16h xa[4];
#pragma unroll
    for (int kk = 0; kk < 4; ++kk)
      xa[kk] = frag_f16(stg + c16 * H_ + kk * 32 + boff);

    v8f ff[8];
#pragma unroll 1
    for (int t = 0; t < 8; ++t) {
      v8f acc = V8F0;
#pragma unroll
      for (int kk = 0; kk < 4; ++kk)
        acc = wmma16(xa[kk], frag_f16(ff1_h + (t * 16 + c16) * H_ + kk * 32 + boff), acc);
      const float bb = b_ff1[t * 16 + c16];
#pragma unroll
      for (int i = 0; i < 8; ++i) ff[t][i] = fmaxf(acc[i] + bb, 0.f);
    }
#pragma unroll
    for (int t = 0; t < 8; ++t)
#pragma unroll
      for (int i = 0; i < 8; ++i)
        stg[(half8 + i) * H_ + t * 16 + c16] = (_Float16)ff[t][i];
    v16h fa[4];
#pragma unroll
    for (int kk = 0; kk < 4; ++kk)
      fa[kk] = frag_f16(stg + c16 * H_ + kk * 32 + boff);

#pragma unroll 1
    for (int t = 0; t < 8; ++t) {
      v8f acc = V8F0;
#pragma unroll
      for (int kk = 0; kk < 4; ++kk)
        acc = wmma16(fa[kk], frag_f16(ff2_h + (t * 16 + c16) * H_ + kk * 32 + boff), acc);
      const float bb = b_ff2[t * 16 + c16];
#pragma unroll
      for (int i = 0; i < 8; ++i) xc[t][i] += acc[i] + bb;
    }
    layernorm_c(xc, ln2_w, ln2_b, c16);

    // ragged sum-pool over valid rows -> LDS float atomics
#pragma unroll
    for (int t = 0; t < 8; ++t) {
      float pl = 0.f;
#pragma unroll
      for (int i = 0; i < 8; ++i) {
        const int r = rt * 16 + half8 + i;
        pl += (r < len) ? xc[t][i] : 0.f;
      }
      pl += __shfl_xor(pl, 16, 32);  // fold upper-lane rows into lower lanes
      if (lane < 16) atomicAdd(&sPool[t * 16 + lane], pl);
    }
  }
  __syncthreads();

  // unit_fc: relu(W_unit @ pooled + b_unit) -> workspace
  if (threadIdx.x < UNITD_) {
    const int k = threadIdx.x;
    float s = b_unit[k];
#pragma unroll 4
    for (int h = 0; h < H_; ++h) s += W_unit[k * H_ + h] * sPool[h];
    unit_out[bu * UNITD_ + k] = fmaxf(s, 0.f);
  }
}

// Tiny fusion MLP: seq_out = sum_u unit, concat with agg + tod, two relu linears.
__global__ __launch_bounds__(128)
void fusion_kernel(const float* __restrict__ unit_out, const float* __restrict__ x_agg,
                   const float* __restrict__ tod_emb, const int* __restrict__ tod_idx,
                   const float* __restrict__ W_fc1, const float* __restrict__ b_fc1,
                   const float* __restrict__ W_fc2, const float* __restrict__ b_fc2,
                   float* __restrict__ out) {
  __shared__ float fused[FUSED_];
  __shared__ float h1[H_];
  const int b = blockIdx.x;
  const int t = threadIdx.x;
  if (t < UNITD_) {
    float s = 0.f;
#pragma unroll 4
    for (int u = 0; u < U_; ++u) s += unit_out[(b * U_ + u) * UNITD_ + t];
    fused[t] = s;
  } else if (t < UNITD_ + AGGD_) {
    fused[t] = x_agg[b * AGGD_ + (t - UNITD_)];
  } else if (t < FUSED_) {
    fused[t] = tod_emb[tod_idx[b] * TODD_ + (t - UNITD_ - AGGD_)];
  }
  __syncthreads();
  float s = b_fc1[t];
#pragma unroll
  for (int j = 0; j < FUSED_; ++j) s += fused[j] * W_fc1[t * FUSED_ + j];
  h1[t] = fmaxf(s, 0.f);
  __syncthreads();
  float o = b_fc2[t];
#pragma unroll 8
  for (int h = 0; h < H_; ++h) o += h1[h] * W_fc2[t * H_ + h];
  out[b * DOUT_ + t] = fmaxf(o, 0.f);
}

extern "C" void kernel_launch(void* const* d_in, const int* in_sizes, int n_in,
                              void* d_out, int out_size, void* d_ws, size_t ws_size,
                              hipStream_t stream) {
  const float* x_seq      = (const float*)d_in[0];
  const float* x_agg      = (const float*)d_in[1];
  const float* tod_emb    = (const float*)d_in[2];
  const float* W_in       = (const float*)d_in[3];
  const float* b_in       = (const float*)d_in[4];
  const float* in_proj_w  = (const float*)d_in[5];
  const float* in_proj_b  = (const float*)d_in[6];
  const float* out_proj_w = (const float*)d_in[7];
  const float* out_proj_b = (const float*)d_in[8];
  const float* ln1_w      = (const float*)d_in[9];
  const float* ln1_b      = (const float*)d_in[10];
  const float* W_ff1      = (const float*)d_in[11];
  const float* b_ff1      = (const float*)d_in[12];
  const float* W_ff2      = (const float*)d_in[13];
  const float* b_ff2      = (const float*)d_in[14];
  const float* ln2_w      = (const float*)d_in[15];
  const float* ln2_b      = (const float*)d_in[16];
  const float* W_unit     = (const float*)d_in[17];
  const float* b_unit     = (const float*)d_in[18];
  const float* W_fc1      = (const float*)d_in[19];
  const float* b_fc1      = (const float*)d_in[20];
  const float* W_fc2      = (const float*)d_in[21];
  const float* b_fc2      = (const float*)d_in[22];
  const int*   lengths    = (const int*)d_in[23];
  const int*   tod_idx    = (const int*)d_in[24];

  // Workspace layout (all 16B-aligned):
  //   [0, 64KB)   unit_out: [B*U][16] f32
  //   [64KB, ..)  f16 weights: in_proj(49152) | out_proj(16384) | ff1(16384) | ff2(16384) | win_pad(4096)
  float*    unit_out   = (float*)d_ws;
  _Float16* wf16       = (_Float16*)((char*)d_ws + 64 * 1024);
  _Float16* in_proj_h  = wf16;
  _Float16* out_proj_h = wf16 + 3 * H_ * H_;
  _Float16* ff1_h      = out_proj_h + H_ * H_;
  _Float16* ff2_h      = ff1_h + H_ * H_;
  _Float16* win_h      = ff2_h + H_ * H_;

  // One-shot weight conversion (stream-ordered before the encoder).
  cvt_f16_kernel<<<dim3((3 * H_ * H_ + 255) / 256), dim3(256), 0, stream>>>(in_proj_w, in_proj_h, 3 * H_ * H_);
  cvt_f16_kernel<<<dim3((H_ * H_ + 255) / 256), dim3(256), 0, stream>>>(out_proj_w, out_proj_h, H_ * H_);
  cvt_f16_kernel<<<dim3((H_ * H_ + 255) / 256), dim3(256), 0, stream>>>(W_ff1, ff1_h, H_ * H_);
  cvt_f16_kernel<<<dim3((H_ * H_ + 255) / 256), dim3(256), 0, stream>>>(W_ff2, ff2_h, H_ * H_);
  pad_win_kernel<<<dim3((H_ * 32 + 255) / 256), dim3(256), 0, stream>>>(W_in, win_h);

  encoder_kernel<<<dim3(B_ * U_), dim3(256), 0, stream>>>(
      x_seq, win_h, b_in, in_proj_h, in_proj_b, out_proj_h, out_proj_b,
      ln1_w, ln1_b, ff1_h, b_ff1, ff2_h, b_ff2, ln2_w, ln2_b,
      W_unit, b_unit, lengths, unit_out);

  fusion_kernel<<<dim3(B_), dim3(DOUT_), 0, stream>>>(
      unit_out, x_agg, tod_emb, tod_idx, W_fc1, b_fc1, W_fc2, b_fc2,
      (float*)d_out);
}
